// lstmDecoder_feedback_68624987456245
// MI455X (gfx1250) — compile-verified, matrix operand order checked by reading, not run on hardware
//
#include <hip/hip_runtime.h>

typedef __attribute__((ext_vector_type(16))) _Float16 v16h;
typedef __attribute__((ext_vector_type(8)))  _Float16 v8h;
typedef __attribute__((ext_vector_type(8)))  float    v8f;

#define B_   4096
#define DZ_  512
#define DIM_ 63
#define T_   120
#define H_   1024
#define G4_  4096   // 4*H

static __device__ __forceinline__ float sigmoidf_(float x) {
  return 1.0f / (1.0f + __expf(-x));
}

// Load a 16x32 f16 operand fragment (rows = M for A, rows = N for B since we
// compute A @ W^T with W row-major [N,K]).  CDNA5 16-bit operand layout:
// lane L holds row (L&15); VGPR 0..3 hold K=kb..kb+7, VGPR 4..7 hold
// K=16+kb..16+kb+7, kb=(L>>4)*8.  Works for global or LDS pointers (the
// address space is inferred after inlining -> ds_load_b128 for shared).
static __device__ __forceinline__ v16h load_frag_f16(const _Float16* base,
                                                     int ld, int row, int k0, int lane) {
  const int r  = lane & 15;
  const int kb = (lane >> 4) << 3;
  const _Float16* p0 = base + (size_t)(row + r) * ld + (k0 + kb);
  v8h lo = *(const v8h*)(p0);        // 16B aligned
  v8h hi = *(const v8h*)(p0 + 16);
  v16h f;
#pragma unroll
  for (int i = 0; i < 8; ++i) { f[i] = lo[i]; f[8 + i] = hi[i]; }
  return f;
}

// ---------------------------------------------------------------------------
// Generic C = act(A[f16, MxK] @ Bw[f16, NxK]^T + bias (+bias2)).
// Wave tile 16Mx64N (4 WMMA accumulators); block = 8 waves = 64M x 128N.
// Output: optional f32 matrix, optional f16 matrix, optional f32 [B,T,DIM]
// slice (first out2_ncut columns).
// ---------------------------------------------------------------------------
__global__ __launch_bounds__(256) void gemm_f16_kernel(
    const _Float16* __restrict__ A, const _Float16* __restrict__ Bw,
    const float* __restrict__ bias, const float* __restrict__ bias2,
    float* __restrict__ Cf32, _Float16* __restrict__ Cf16,
    int M, int N, int K, int bias_n, int do_relu,
    float* __restrict__ out2, int out2_rowstride, int out2_base, int out2_ncut)
{
  const int lane = threadIdx.x & 31;
  const int wave = threadIdx.x >> 5;
  const int mw = wave & 3, nw = wave >> 2;
  const int m_base = blockIdx.y * 64 + mw * 16;
  const int n_base = blockIdx.x * 128 + nw * 64;
  if (n_base >= N) return;   // wave-uniform: EXEC stays all-ones for WMMA

  v8f acc[4];
#pragma unroll
  for (int j = 0; j < 4; ++j)
#pragma unroll
    for (int i = 0; i < 8; ++i) acc[j][i] = 0.0f;

  for (int k0 = 0; k0 < K; k0 += 32) {
    v16h a = load_frag_f16(A, K, m_base, k0, lane);
#pragma unroll
    for (int j = 0; j < 4; ++j) {
      v16h b = load_frag_f16(Bw, K, n_base + 16 * j, k0, lane);
      acc[j] = __builtin_amdgcn_wmma_f32_16x16x32_f16(
          false, a, false, b, (short)0, acc[j], false, false);
    }
  }

  const int rsel  = (lane >> 4) << 3;
  const int nlane = lane & 15;
#pragma unroll
  for (int j = 0; j < 4; ++j) {
    const int ncol = n_base + 16 * j + nlane;
    float bv = 0.0f;
    if (bias  != nullptr && ncol < bias_n) bv += bias[ncol];
    if (bias2 != nullptr && ncol < bias_n) bv += bias2[ncol];
#pragma unroll
    for (int i = 0; i < 8; ++i) {
      const int mrow = m_base + i + rsel;
      float v = acc[j][i] + bv;
      if (do_relu) v = fmaxf(v, 0.0f);
      const size_t cidx = (size_t)mrow * N + ncol;
      if (Cf32 != nullptr) Cf32[cidx] = v;
      if (Cf16 != nullptr) Cf16[cidx] = (_Float16)v;
      if (out2 != nullptr && ncol < out2_ncut)
        out2[(size_t)mrow * out2_rowstride + out2_base + ncol] = v;
    }
  }
}

// ---------------------------------------------------------------------------
// Fused LSTM step.  Block = 8 waves covering a 32M x 128N tile of h.
// The 32x1024 f16 tile of h_in (shared A-operand of every wave in the block)
// is staged once into LDS (64 KB) and A fragments are read via ds_load.
// Each wave owns a 32M x 16N h-tile: 8 accumulators (4 gates x 2 m-subtiles),
// so every W_hh B-fragment is reused for 2 WMMAs.  Gate g of hidden column j
// lives at gates column g*H + j, so the full cell update is local to the wave:
// start from loop-invariant gates_x, add pose @ W_ihp^T and h @ W_hh^T, apply
// sigmoid/tanh, update c (f32, in place) and write h_out (f16).
// ---------------------------------------------------------------------------
__global__ __launch_bounds__(256) void lstm_step_kernel(
    const float* __restrict__ gates_x, const _Float16* __restrict__ pose,
    const _Float16* __restrict__ h_in, const _Float16* __restrict__ Whh,
    const _Float16* __restrict__ Wihp, float* __restrict__ c,
    _Float16* __restrict__ h_out)
{
  __shared__ _Float16 sh[32 * H_];   // 64 KB: 32 batch rows x K=1024 of h_in

  const int lane = threadIdx.x & 31;
  const int wave = threadIdx.x >> 5;
  // grid: (B/32) * (H/128) blocks; 8 n-blocks per 32-row m-group
  const int m_base = (blockIdx.x >> 3) * 32;
  const int n_base = (blockIdx.x & 7) * 128 + wave * 16;
  const int rsel  = (lane >> 4) << 3;
  const int nlane = lane & 15;

  // Cooperative stage of h_in[m_base:m_base+32, :] -> LDS (v8h granularity).
  const int tid = threadIdx.x;
#pragma unroll
  for (int it = 0; it < 16; ++it) {
    const int idx = it * 2048 + tid * 8;       // 256 thr * 8 f16 = 2048/iter
    const int row = idx >> 10, col = idx & (H_ - 1);
    *(v8h*)(&sh[idx]) = *(const v8h*)(h_in + (size_t)(m_base + row) * H_ + col);
  }

  v8f acc[4][2];
#pragma unroll
  for (int g = 0; g < 4; ++g)
#pragma unroll
    for (int s = 0; s < 2; ++s)
#pragma unroll
      for (int i = 0; i < 8; ++i) {
        const int mrow = m_base + s * 16 + i + rsel;
        acc[g][s][i] = gates_x[(size_t)mrow * G4_ + g * H_ + n_base + nlane];
      }

  __syncthreads();

  for (int k0 = 0; k0 < H_; k0 += 32) {
    v16h a0 = load_frag_f16(sh, H_, 0,  k0, lane);   // LDS reads
    v16h a1 = load_frag_f16(sh, H_, 16, k0, lane);
#pragma unroll
    for (int g = 0; g < 4; ++g) {
      v16h b = load_frag_f16(Whh, H_, g * H_ + n_base, k0, lane);
      acc[g][0] = __builtin_amdgcn_wmma_f32_16x16x32_f16(
          false, a0, false, b, (short)0, acc[g][0], false, false);
      acc[g][1] = __builtin_amdgcn_wmma_f32_16x16x32_f16(
          false, a1, false, b, (short)0, acc[g][1], false, false);
    }
  }

#pragma unroll
  for (int k0 = 0; k0 < 64; k0 += 32) {   // pose contribution, DIM padded to 64
    v16h a0 = load_frag_f16(pose, 64, m_base,      k0, lane);
    v16h a1 = load_frag_f16(pose, 64, m_base + 16, k0, lane);
#pragma unroll
    for (int g = 0; g < 4; ++g) {
      v16h b = load_frag_f16(Wihp, 64, g * H_ + n_base, k0, lane);
      acc[g][0] = __builtin_amdgcn_wmma_f32_16x16x32_f16(
          false, a0, false, b, (short)0, acc[g][0], false, false);
      acc[g][1] = __builtin_amdgcn_wmma_f32_16x16x32_f16(
          false, a1, false, b, (short)0, acc[g][1], false, false);
    }
  }

#pragma unroll
  for (int s = 0; s < 2; ++s)
#pragma unroll
    for (int i = 0; i < 8; ++i) {
      const int mrow = m_base + s * 16 + i + rsel;
      const size_t idx = (size_t)mrow * H_ + n_base + nlane;
      const float cv = c[idx];
      const float ig = sigmoidf_(acc[0][s][i]);
      const float fg = sigmoidf_(acc[1][s][i]);
      const float gg = tanhf(acc[2][s][i]);
      const float og = sigmoidf_(acc[3][s][i]);
      const float cn = fg * cv + ig * gg;
      const float hn = og * tanhf(cn);
      c[idx] = cn;
      h_out[idx] = (_Float16)hn;
    }
}

// f32 -> f16 conversion with zero padding (rows and/or cols).
__global__ void cvt_pad_kernel(_Float16* __restrict__ dst, const float* __restrict__ src,
                               int src_ld, int col_off, int rows_valid, int cols_valid,
                               int Kp, int total) {
  int idx = blockIdx.x * blockDim.x + threadIdx.x;
  if (idx >= total) return;
  int r = idx / Kp, k = idx % Kp;
  float v = (r < rows_valid && k < cols_valid) ? src[(size_t)r * src_ld + col_off + k] : 0.0f;
  dst[idx] = (_Float16)v;
}

__global__ void zero_f32_kernel(float* __restrict__ p, int n) {
  int i = blockIdx.x * blockDim.x + threadIdx.x;
  if (i < n) p[i] = 0.0f;
}

__global__ void zero_f16_kernel(_Float16* __restrict__ p, int n) {
  int i = blockIdx.x * blockDim.x + threadIdx.x;
  if (i < n) p[i] = (_Float16)0.0f;
}

extern "C" void kernel_launch(void* const* d_in, const int* in_sizes, int n_in,
                              void* d_out, int out_size, void* d_ws, size_t ws_size,
                              hipStream_t stream) {
  (void)in_sizes; (void)n_in; (void)out_size; (void)ws_size;
  const float* x     = (const float*)d_in[0];
  const float* W_lin = (const float*)d_in[1];
  const float* b_lin = (const float*)d_in[2];
  const float* W_p1  = (const float*)d_in[3];
  const float* b_p1  = (const float*)d_in[4];
  const float* W_p2  = (const float*)d_in[5];
  const float* b_p2  = (const float*)d_in[6];
  const float* W_ih  = (const float*)d_in[7];
  const float* b_ih  = (const float*)d_in[8];
  const float* W_hh  = (const float*)d_in[9];
  const float* b_hh  = (const float*)d_in[10];
  const float* W_1   = (const float*)d_in[11];
  const float* b_1   = (const float*)d_in[12];
  const float* W_2   = (const float*)d_in[13];
  const float* b_2   = (const float*)d_in[14];
  // d_in[15] = input_len (device scalar); loop length is fixed at T_=120.
  float* out = (float*)d_out;

  char* ws = (char*)d_ws;
  size_t off = 0;
  auto take = [&](size_t bytes) -> char* {
    char* p = ws + off;
    off = (off + bytes + 255) & ~(size_t)255;
    return p;
  };
  float*    gates_x = (float*)take((size_t)B_ * G4_ * 4);  // loop-invariant
  float*    cbuf    = (float*)take((size_t)B_ * H_ * 4);   // cell state, f32
  _Float16* hA16    = (_Float16*)take((size_t)B_ * H_ * 2);
  _Float16* hB16    = (_Float16*)take((size_t)B_ * H_ * 2);
  _Float16* pose16  = (_Float16*)take((size_t)B_ * 64 * 2);
  _Float16* t16     = (_Float16*)take((size_t)B_ * DZ_ * 2);
  _Float16* x116    = (_Float16*)take((size_t)B_ * DZ_ * 2);
  _Float16* x16     = (_Float16*)take((size_t)B_ * DZ_ * 2);
  _Float16* W_lin16 = (_Float16*)take((size_t)512 * 512 * 2);
  _Float16* W_p116  = (_Float16*)take((size_t)512 * 512 * 2);
  _Float16* W_p216  = (_Float16*)take((size_t)64 * 512 * 2);
  _Float16* W_ihx16 = (_Float16*)take((size_t)4096 * 512 * 2);
  _Float16* W_ihp16 = (_Float16*)take((size_t)4096 * 64 * 2);
  _Float16* W_hh16  = (_Float16*)take((size_t)4096 * 1024 * 2);
  _Float16* W_116   = (_Float16*)take((size_t)512 * 1024 * 2);
  _Float16* W_216   = (_Float16*)take((size_t)64 * 512 * 2);

  auto cvt = [&](_Float16* dst, const float* src, int src_ld, int col_off,
                 int rows_valid, int cols_valid, int Kp, int rows_tot) {
    int total = rows_tot * Kp;
    cvt_pad_kernel<<<(total + 255) / 256, 256, 0, stream>>>(
        dst, src, src_ld, col_off, rows_valid, cols_valid, Kp, total);
  };
  cvt(x16,     x,     512, 0, B_, 512, 512, B_);       // activations -> f16
  cvt(W_lin16, W_lin, 512, 0, 512, 512, 512, 512);
  cvt(W_p116,  W_p1,  512, 0, 512, 512, 512, 512);
  cvt(W_p216,  W_p2,  512, 0, DIM_, 512, 512, 64);     // pad row 63 -> 0
  cvt(W_ihx16, W_ih,  575, 0, 4096, 512, 512, 4096);   // W_ih[:, :512]
  cvt(W_ihp16, W_ih,  575, 512, 4096, DIM_, 64, 4096); // W_ih[:, 512:575], pad
  cvt(W_hh16,  W_hh,  1024, 0, 4096, 1024, 1024, 4096);
  cvt(W_116,   W_1,   1024, 0, 512, 1024, 1024, 512);
  cvt(W_216,   W_2,   512, 0, DIM_, 512, 512, 64);     // pad row 63 -> 0

  {
    int n = B_ * H_;
    zero_f16_kernel<<<(n + 255) / 256, 256, 0, stream>>>(hA16, n);
    zero_f32_kernel<<<(n + 255) / 256, 256, 0, stream>>>(cbuf, n);
  }

  auto gemm = [&](const _Float16* Ax, const _Float16* Bx, const float* bv1,
                  const float* bv2, float* C32, _Float16* C16,
                  int M, int N, int K, int bias_n, int relu,
                  float* o2, int o2rs, int o2b, int o2n) {
    dim3 grid((N + 127) / 128, (M + 63) / 64);
    gemm_f16_kernel<<<grid, 256, 0, stream>>>(Ax, Bx, bv1, bv2, C32, C16,
                                              M, N, K, bias_n, relu,
                                              o2, o2rs, o2b, o2n);
  };

  // Prologue: x1 = relu(x@W_lin^T+b); pose0; gates_x = x1@W_ihx^T+b_ih+b_hh.
  gemm(x16,  W_lin16, b_lin, nullptr, nullptr, x116,   B_, 512, 512, 512, 1,
       nullptr, 0, 0, 0);
  gemm(x116, W_p116,  b_p1,  nullptr, nullptr, t16,    B_, 512, 512, 512, 1,
       nullptr, 0, 0, 0);
  gemm(t16,  W_p216,  b_p2,  nullptr, nullptr, pose16, B_, 64, 512, DIM_, 0,
       nullptr, 0, 0, 0);
  gemm(x116, W_ihx16, b_ih,  b_hh,    gates_x, nullptr, B_, G4_, 512, G4_, 0,
       nullptr, 0, 0, 0);

  // Recurrence: 120 sequential steps; h double-buffered (f16), c in place.
  const int lstm_blocks = (B_ / 32) * (H_ / 128);   // 128 * 8 = 1024
  for (int s = 0; s < T_; ++s) {
    _Float16* h_r = (s & 1) ? hB16 : hA16;
    _Float16* h_w = (s & 1) ? hA16 : hB16;
    lstm_step_kernel<<<lstm_blocks, 256, 0, stream>>>(
        gates_x, pose16, h_r, W_hh16, W_ihp16, cbuf, h_w);
    gemm(h_w, W_116, b_1, nullptr, nullptr, t16, B_, 512, 1024, 512, 1,
         nullptr, 0, 0, 0);
    // pose update also writes the output slice out[:, s, :]
    gemm(t16, W_216, b_2, nullptr, nullptr, pose16, B_, 64, 512, DIM_, 0,
         out, T_ * DIM_, s * DIM_, DIM_);
  }
}
